// LSTM_69879117906487
// MI455X (gfx1250) — matrix operand order repaired and checked by reading, not compiled, not run on hardware
//
#include <hip/hip_runtime.h>
#include <hip/hip_bf16.h>

// MI455X (gfx1250, wave32) LSTM+MLP pipeline:
//   - bf16 WMMA (v_wmma_f32_16x16x32_bf16) for all GEMMs, f32 accumulate
//   - TDM tensor_load_to_lds (TENSORcnt) for 2D weight/activation tiles
//   - async global->LDS copies (ASYNCcnt) for remaining staging
//   - f32 VALU for gate nonlinearities

typedef __attribute__((ext_vector_type(8)))  float          v8f;
typedef __attribute__((ext_vector_type(16))) __bf16         v16bf;
typedef __attribute__((ext_vector_type(8)))  unsigned short v8us;
typedef __attribute__((ext_vector_type(16))) unsigned short v16us;
typedef __attribute__((ext_vector_type(4)))  unsigned int   u32x4;
typedef __attribute__((ext_vector_type(4)))  int            i32x4;
typedef __attribute__((ext_vector_type(8)))  int            i32x8;

__device__ __forceinline__ unsigned short f2bf(float f) {
  union { float f; unsigned u; } v; v.f = f;
  unsigned r = v.u + 0x7FFFu + ((v.u >> 16) & 1u);   // RNE
  return (unsigned short)(r >> 16);
}

__device__ __forceinline__ v16bf frag16(v8us lo, v8us hi) {
  v16us t;
#pragma unroll
  for (int i = 0; i < 8; ++i) { t[i] = lo[i]; t[8 + i] = hi[i]; }
  return __builtin_bit_cast(v16bf, t);
}

// ---- gfx1250 async global->LDS copy (16 bytes per active lane) ----
__device__ __forceinline__ void async_cp16(const void* gsrc, void* ldst) {
  unsigned lds_addr = (unsigned)(unsigned long long)ldst;
  asm volatile("global_load_async_to_lds_b128 %0, %1, off"
               :: "v"(lds_addr), "v"(gsrc)
               : "memory");
}

__device__ __forceinline__ void async_wait0() {
#if __has_builtin(__builtin_amdgcn_s_wait_asynccnt)
  __builtin_amdgcn_s_wait_asynccnt(0);
#else
  asm volatile("s_wait_asynccnt 0x0" ::: "memory");
#endif
}

// ---- gfx1250 TDM: 2D bf16 tile, global -> LDS, with LDS row padding ----
// D# per CDNA5 ISA ch.8: group0 = {count, lds_addr, global_addr, type=2},
// group1 = {data_size=2B, pad(4 dwords per 16 dwords), dims/strides}.
// Tile is tile_w x tile_h elements from a row-major tensor (stride_w elems);
// with pad enabled each 32-element row lands on a 40-element LDS stride.
struct TdmG1 { unsigned long long q0, q1, q2, q3; };

__device__ __forceinline__ void tdm_load_tile_bf16(
    const void* gsrc, void* ldst, unsigned tensor_w, unsigned tensor_h,
    unsigned stride_w, unsigned tile_w, unsigned tile_h) {
  unsigned long long ga = (unsigned long long)gsrc;
  unsigned lds = (unsigned)(unsigned long long)ldst;

  u32x4 g0;
  g0[0] = 1u;                                              // count=1, load, user
  g0[1] = lds;                                             // lds_addr
  g0[2] = (unsigned)ga;                                    // global_addr[31:0]
  g0[3] = (unsigned)((ga >> 32) & 0x01FFFFFFu) | (2u << 30);  // addr[56:32]|type=2

  // group1 bitfields (bit positions per ISA 8.4)
  unsigned w0 = (1u << 16)            // data_size = 2 bytes
              | (1u << 20)            // pad_enable
              | (3u << 22)            // pad_interval: 16 dwords (one 32-elem row)
              | (3u << 25);           // pad_amount:   4 dwords  (8 bf16 pad)
  TdmG1 g1;
  g1.q0 = (unsigned long long)w0
        | ((unsigned long long)(tensor_w & 0xFFFFu) << 48);        // dim0[15:0]
  g1.q1 = (unsigned long long)(tensor_w >> 16)                     // dim0[31:16]
        | ((unsigned long long)tensor_h << 16)                     // dim1
        | ((unsigned long long)(tile_w & 0xFFFFu) << 48);          // tile_dim0
  g1.q2 = (unsigned long long)(tile_h & 0xFFFFu)                   // tile_dim1
        | ((unsigned long long)stride_w << 32);                    // dim0_stride lo32
  g1.q3 = 0ull;                                                    // dim1_stride unused (2D)

  i32x8 g1v = __builtin_bit_cast(i32x8, g1);
  i32x4 z4 = {};
#if __has_builtin(__builtin_amdgcn_tensor_load_to_lds)
#if __has_include(<hip/amd_detail/amd_gfx1250_TDM.h>)
  i32x8 z8 = {};
  __builtin_amdgcn_tensor_load_to_lds(g0, g1v, z4, z4, z8, 0);   // clang-23 6-arg
#else
  __builtin_amdgcn_tensor_load_to_lds(g0, g1v, z4, z4, 0);       // ROCm 7.2 5-arg
#endif
#endif
}

__device__ __forceinline__ void tensor_wait0() {
#if __has_builtin(__builtin_amdgcn_s_wait_tensorcnt)
  __builtin_amdgcn_s_wait_tensorcnt(0);
#else
  asm volatile("s_wait_tensorcnt 0x0" ::: "memory");
#endif
}

// ---------------------------------------------------------------------------
// Elementwise prep kernels
// ---------------------------------------------------------------------------
__global__ void k_cvt(const float* __restrict__ src, unsigned short* __restrict__ dst, int n) {
  int i = blockIdx.x * 256 + threadIdx.x;
  if (i < n) dst[i] = f2bf(src[i]);
}

__global__ void k_bias_init(const float* __restrict__ bih, const float* __restrict__ bhh,
                            float* __restrict__ bias, unsigned short* __restrict__ hbf,
                            float* __restrict__ cbuf) {
  int i = blockIdx.x * 256 + threadIdx.x;      // grid covers 65536
  if (i < 1024) bias[i] = bih[i] + bhh[i];
  if (i < 65536) { hbf[i] = 0; cbuf[i] = 0.0f; }
}

// ---------------------------------------------------------------------------
// GEMM1: xg[m, n] = bias[n] + sum_k x[m,k] * Wih[n,k]
//   M = 65536 (T*B), N = 1024 (4H), K = 1024.  A converted f32->bf16 inline,
//   B staged by TDM.  Tile 128x64, BK=32, 8 waves, wave = 32x32.
// ---------------------------------------------------------------------------
__global__ __launch_bounds__(256) void k_gemm_xg(
    const float* __restrict__ x, const unsigned short* __restrict__ Wih,
    const float* __restrict__ bias, float* __restrict__ xg) {
  __shared__ unsigned short sA[128 * 40];   // 128 rows x 32 bf16 (+8 pad)
  __shared__ unsigned short sB[64 * 40];    // 64 cols  x 32 bf16 (+8 pad)

  const int tid = threadIdx.x, lane = tid & 31, wid = tid >> 5;
  const int hl = lane >> 4, l16 = lane & 15;
  const int wm = (wid & 3) * 32, wn = (wid >> 2) * 32;
  const int m0 = blockIdx.x * 128, n0 = blockIdx.y * 64;

  const int arow = tid >> 1, akp = (tid & 1) * 16;   // A fill: 16 f32 each

  v8f acc[2][2] = {};

  for (int k0 = 0; k0 < 1024; k0 += 32) {
    // --- stage B: Wih 64x32 tile via TDM (wave 0 only, wave-uniform) ---
    if (wid == 0)
      tdm_load_tile_bf16(Wih + (size_t)n0 * 1024 + k0, sB,
                         1024, 1024, 1024, 32, 64);
    // --- stage A: x f32 -> bf16 into LDS (needs VALU convert) ---
    const float4* gx4 = (const float4*)(x + (size_t)(m0 + arow) * 1024 + k0 + akp);
    float4 f0 = gx4[0], f1 = gx4[1], f2 = gx4[2], f3 = gx4[3];
    if (k0 + 32 < 1024) __builtin_prefetch((const char*)gx4 + 128, 0, 1);
    v8us a0, a1;
    a0[0]=f2bf(f0.x); a0[1]=f2bf(f0.y); a0[2]=f2bf(f0.z); a0[3]=f2bf(f0.w);
    a0[4]=f2bf(f1.x); a0[5]=f2bf(f1.y); a0[6]=f2bf(f1.z); a0[7]=f2bf(f1.w);
    a1[0]=f2bf(f2.x); a1[1]=f2bf(f2.y); a1[2]=f2bf(f2.z); a1[3]=f2bf(f2.w);
    a1[4]=f2bf(f3.x); a1[5]=f2bf(f3.y); a1[6]=f2bf(f3.z); a1[7]=f2bf(f3.w);
    *(v8us*)&sA[arow * 40 + akp]     = a0;
    *(v8us*)&sA[arow * 40 + akp + 8] = a1;
    if (wid == 0) tensor_wait0();
    __syncthreads();

    // --- fragments per ISA 7.12.2 layout ---
    v16bf afr[2], bfr[2];
#pragma unroll
    for (int mi = 0; mi < 2; ++mi) {
      const unsigned short* ap = &sA[(wm + mi * 16 + l16) * 40];
      afr[mi] = frag16(*(const v8us*)(ap + hl * 8), *(const v8us*)(ap + 16 + hl * 8));
    }
#pragma unroll
    for (int ni = 0; ni < 2; ++ni) {
      const unsigned short* bp = &sB[(wn + ni * 16 + l16) * 40 + hl * 8];
      bfr[ni] = frag16(*(const v8us*)bp, *(const v8us*)(bp + 16));
    }
#pragma unroll
    for (int mi = 0; mi < 2; ++mi)
#pragma unroll
      for (int ni = 0; ni < 2; ++ni)
        acc[mi][ni] = __builtin_amdgcn_wmma_f32_16x16x32_bf16(
            false, afr[mi], false, bfr[ni], (short)0, acc[mi][ni], false, false);
    __syncthreads();
  }

#pragma unroll
  for (int mi = 0; mi < 2; ++mi)
#pragma unroll
    for (int ni = 0; ni < 2; ++ni) {
      int gn = n0 + wn + ni * 16 + l16;
      float bv = bias[gn];
#pragma unroll
      for (int r = 0; r < 8; ++r) {
        int gm = m0 + wm + mi * 16 + hl * 8 + r;
        xg[(size_t)gm * 1024 + gn] = acc[mi][ni][r] + bv;
      }
    }
}

// ---------------------------------------------------------------------------
// Recurrent GEMM: z[b, n] = xg_t[b, n] + sum_k h[b,k] * Whh[n,k]
//   M = 256 (batch), N = 1024 (4H), K = 256 (H).
//   A staged async-to-LDS, B staged by TDM.
// ---------------------------------------------------------------------------
__global__ __launch_bounds__(256) void k_gemm_z(
    const unsigned short* __restrict__ hbf, const unsigned short* __restrict__ Whh,
    const float* __restrict__ xg_t, float* __restrict__ z) {
  __shared__ unsigned short sA[128 * 40];
  __shared__ unsigned short sB[64 * 40];

  const int tid = threadIdx.x, lane = tid & 31, wid = tid >> 5;
  const int hl = lane >> 4, l16 = lane & 15;
  const int wm = (wid & 3) * 32, wn = (wid >> 2) * 32;
  const int m0 = blockIdx.x * 128, n0 = blockIdx.y * 64;

  const int arow = tid >> 1, akp = (tid & 1) * 16;

  v8f acc[2][2] = {};

  for (int k0 = 0; k0 < 256; k0 += 32) {
    if (wid == 0)
      tdm_load_tile_bf16(Whh + (size_t)n0 * 256 + k0, sB,
                         256, 1024, 256, 32, 64);
    const unsigned short* gh = hbf + (size_t)(m0 + arow) * 256 + k0 + akp;
    async_cp16(gh,     &sA[arow * 40 + akp]);
    async_cp16(gh + 8, &sA[arow * 40 + akp + 8]);
    if (wid == 0) tensor_wait0();
    async_wait0();
    __syncthreads();

    v16bf afr[2], bfr[2];
#pragma unroll
    for (int mi = 0; mi < 2; ++mi) {
      const unsigned short* ap = &sA[(wm + mi * 16 + l16) * 40];
      afr[mi] = frag16(*(const v8us*)(ap + hl * 8), *(const v8us*)(ap + 16 + hl * 8));
    }
#pragma unroll
    for (int ni = 0; ni < 2; ++ni) {
      const unsigned short* bp = &sB[(wn + ni * 16 + l16) * 40 + hl * 8];
      bfr[ni] = frag16(*(const v8us*)bp, *(const v8us*)(bp + 16));
    }
#pragma unroll
    for (int mi = 0; mi < 2; ++mi)
#pragma unroll
      for (int ni = 0; ni < 2; ++ni)
        acc[mi][ni] = __builtin_amdgcn_wmma_f32_16x16x32_bf16(
            false, afr[mi], false, bfr[ni], (short)0, acc[mi][ni], false, false);
    __syncthreads();
  }

#pragma unroll
  for (int mi = 0; mi < 2; ++mi)
#pragma unroll
    for (int ni = 0; ni < 2; ++ni) {
      int gn = n0 + wn + ni * 16 + l16;
#pragma unroll
      for (int r = 0; r < 8; ++r) {
        int gm = m0 + wm + mi * 16 + hl * 8 + r;
        size_t off = (size_t)gm * 1024 + gn;
        z[off] = acc[mi][ni][r] + xg_t[off];
      }
    }
}

// ---------------------------------------------------------------------------
// Gate nonlinearity + state update (f32), emits next-h (bf16) and relu(h) (bf16)
// ---------------------------------------------------------------------------
__global__ void k_gates(const float* __restrict__ z, float* __restrict__ c,
                        unsigned short* __restrict__ hbf, unsigned short* __restrict__ hseq_t) {
  int idx = blockIdx.x * 256 + threadIdx.x;   // 65536 = B*H
  int b = idx >> 8, hh = idx & 255;
  const float* zr = z + (size_t)b * 1024;
  float gi = zr[hh], gf = zr[256 + hh], gg = zr[512 + hh], go = zr[768 + hh];
  gi = 1.0f / (1.0f + __expf(-gi));
  gf = 1.0f / (1.0f + __expf(-gf));
  gg = tanhf(gg);
  go = 1.0f / (1.0f + __expf(-go));
  float cn = gf * c[idx] + gi * gg;
  c[idx] = cn;
  float h = go * tanhf(cn);
  hbf[idx] = f2bf(h);
  hseq_t[idx] = f2bf(fmaxf(h, 0.0f));
}

// ---------------------------------------------------------------------------
// Fused FC1+FC2: out = relu(hseq @ W1^T + b1) @ W2^T + b2
//   Stage1: M=128/block, N1=128, K1=256 (WMMA), TDM-staged tiles.
//   Stage2: K2=128, N2=8 padded to 16 with zero rows in LDS W2 tile (WMMA).
// ---------------------------------------------------------------------------
__global__ __launch_bounds__(256) void k_fc(
    const unsigned short* __restrict__ hseq, const unsigned short* __restrict__ W1b,
    const float* __restrict__ b1, const unsigned short* __restrict__ W2b,
    const float* __restrict__ b2, float* __restrict__ out) {
  __shared__ unsigned short sH[128 * 40];
  __shared__ unsigned short sW[128 * 40];
  __shared__ unsigned short sF[128 * 136];   // fc1 activations, bf16, K-contiguous
  __shared__ unsigned short sW2[16 * 136];   // W2 padded 8->16 rows

  const int tid = threadIdx.x, lane = tid & 31, wid = tid >> 5;
  const int hl = lane >> 4, l16 = lane & 15;
  const int m0 = blockIdx.x * 128;

  { // W2 pad tile: waves 0-3 async-copy rows 0..7, waves 4-7 zero rows 8..15
    int row = tid >> 4, cp = (tid & 15) * 8;
    if (row < 8) {
      async_cp16(W2b + row * 128 + cp, &sW2[row * 136 + cp]);
    } else {
      v8us zz = {0, 0, 0, 0, 0, 0, 0, 0};
      *(v8us*)&sW2[row * 136 + cp] = zz;
    }
  }

  // ---- stage 1: hseq @ W1^T ----
  const int wm = (wid & 3) * 32, wn = (wid >> 2) * 64;
  v8f acc[2][4] = {};

  for (int k0 = 0; k0 < 256; k0 += 32) {
    if (wid == 0) {   // two TDM 2D tiles: activations + weights
      tdm_load_tile_bf16(hseq + (size_t)m0 * 256 + k0, sH,
                         256, 65536, 256, 32, 128);
      tdm_load_tile_bf16(W1b + k0, sW,
                         256, 128, 256, 32, 128);
      tensor_wait0();
    }
    async_wait0();
    __syncthreads();

    v16bf afr[2], bfr[4];
#pragma unroll
    for (int mi = 0; mi < 2; ++mi) {
      const unsigned short* ap = &sH[(wm + mi * 16 + l16) * 40];
      afr[mi] = frag16(*(const v8us*)(ap + hl * 8), *(const v8us*)(ap + 16 + hl * 8));
    }
#pragma unroll
    for (int ni = 0; ni < 4; ++ni) {
      const unsigned short* bp = &sW[(wn + ni * 16 + l16) * 40 + hl * 8];
      bfr[ni] = frag16(*(const v8us*)bp, *(const v8us*)(bp + 16));
    }
#pragma unroll
    for (int mi = 0; mi < 2; ++mi)
#pragma unroll
      for (int ni = 0; ni < 4; ++ni)
        acc[mi][ni] = __builtin_amdgcn_wmma_f32_16x16x32_bf16(
            false, afr[mi], false, bfr[ni], (short)0, acc[mi][ni], false, false);
    __syncthreads();
  }

  // bias + relu -> bf16 LDS tile
#pragma unroll
  for (int mi = 0; mi < 2; ++mi)
#pragma unroll
    for (int ni = 0; ni < 4; ++ni) {
      int n = wn + ni * 16 + l16;
      float bv = b1[n];
#pragma unroll
      for (int r = 0; r < 8; ++r) {
        int m = wm + mi * 16 + hl * 8 + r;
        sF[m * 136 + n] = f2bf(fmaxf(acc[mi][ni][r] + bv, 0.0f));
      }
    }
  __syncthreads();

  // ---- stage 2: fc1 @ W2^T (N padded to 16) ----
  v8f acc2 = {};
  for (int k0 = 0; k0 < 128; k0 += 32) {
    const unsigned short* ap = &sF[(wid * 16 + l16) * 136 + k0];
    v16bf a = frag16(*(const v8us*)(ap + hl * 8), *(const v8us*)(ap + 16 + hl * 8));
    const unsigned short* bp = &sW2[l16 * 136 + k0 + hl * 8];
    v16bf b = frag16(*(const v8us*)bp, *(const v8us*)(bp + 16));
    acc2 = __builtin_amdgcn_wmma_f32_16x16x32_bf16(false, a, false, b, (short)0, acc2,
                                                   false, false);
  }
  if (l16 < 8) {
    float bb = b2[l16];
#pragma unroll
    for (int r = 0; r < 8; ++r) {
      int gm = m0 + wid * 16 + hl * 8 + r;
      out[(size_t)gm * 8 + l16] = acc2[r] + bb;
    }
  }
}

// ---------------------------------------------------------------------------
extern "C" void kernel_launch(void* const* d_in, const int* in_sizes, int n_in,
                              void* d_out, int out_size, void* d_ws, size_t ws_size,
                              hipStream_t stream) {
  const float* x   = (const float*)d_in[0];
  const float* Wih = (const float*)d_in[1];
  const float* Whh = (const float*)d_in[2];
  const float* bih = (const float*)d_in[3];
  const float* bhh = (const float*)d_in[4];
  const float* W1  = (const float*)d_in[5];
  const float* b1  = (const float*)d_in[6];
  const float* W2  = (const float*)d_in[7];
  const float* b2  = (const float*)d_in[8];
  float* out = (float*)d_out;

  char* p = (char*)d_ws;
  auto carve = [&](size_t bytes) {
    char* r = p; p += (bytes + 255) & ~(size_t)255; return r;
  };
  float*          xg   = (float*)carve(65536ull * 1024 * 4);   // 268 MB
  float*          zbuf = (float*)carve(256 * 1024 * 4);
  float*          cbuf = (float*)carve(65536 * 4);
  unsigned short* hbf  = (unsigned short*)carve(65536 * 2);
  unsigned short* hseq = (unsigned short*)carve(65536ull * 256 * 2);  // 33.5 MB
  unsigned short* WihB = (unsigned short*)carve(1024 * 1024 * 2);
  unsigned short* WhhB = (unsigned short*)carve(1024 * 256 * 2);
  unsigned short* W1B  = (unsigned short*)carve(128 * 256 * 2);
  unsigned short* W2B  = (unsigned short*)carve(8 * 128 * 2);
  float*          bias = (float*)carve(1024 * 4);

  k_cvt<<<(1024 * 1024 + 255) / 256, 256, 0, stream>>>(Wih, WihB, 1024 * 1024);
  k_cvt<<<(1024 * 256 + 255) / 256, 256, 0, stream>>>(Whh, WhhB, 1024 * 256);
  k_cvt<<<(128 * 256 + 255) / 256, 256, 0, stream>>>(W1, W1B, 128 * 256);
  k_cvt<<<(8 * 128 + 255) / 256, 256, 0, stream>>>(W2, W2B, 8 * 128);
  k_bias_init<<<256, 256, 0, stream>>>(bih, bhh, bias, hbf, cbuf);

  dim3 g1(65536 / 128, 1024 / 64);
  k_gemm_xg<<<g1, 256, 0, stream>>>(x, WihB, bias, xg);

  dim3 gz(256 / 128, 1024 / 64);
  for (int t = 0; t < 256; ++t) {
    k_gemm_z<<<gz, 256, 0, stream>>>(hbf, WhhB, xg + (size_t)t * 256 * 1024, zbuf);
    k_gates<<<256, 256, 0, stream>>>(zbuf, cbuf, hbf, hseq + (size_t)t * 65536);
  }

  k_fc<<<65536 / 128, 256, 0, stream>>>(hseq, W1B, b1, W2B, b2, out);
}
